// MambaRecursiveStateAdjustmentV4_41944650613243
// MI455X (gfx1250) — matrix operand order repaired and checked by reading, not compiled, not run on hardware
//
#include <hip/hip_runtime.h>

// ---------------- problem constants ----------------
#define C96   96
#define DI    192
#define DS16  16
#define KDIR  4
#define HH    48
#define LHW   2304           // 48*48
#define NBAT  8
#define NCOL  18432          // NBAT*LHW
#define SEQLEN 4

typedef _Float16 f16x16 __attribute__((ext_vector_type(16)));
typedef _Float16 f16x8  __attribute__((ext_vector_type(8)));
typedef float    f32x8  __attribute__((ext_vector_type(8)));

union FragU { f16x16 v; f16x8 h[2]; };

__device__ __forceinline__ float sigmoidf_(float x) { return 1.f / (1.f + __expf(-x)); }

// =====================================================================
// WMMA GEMM:  Out[M,N] (f32) = A[M,Kpad](f16, zero-padded) * B[K,N]
// B source (template: f32 or f16), plain row-major with row stride bRow.
// Block tile: 32 (M) x 128 (N), 256 threads = 8 waves.
// Wave tile: 16 (M) x 32 (N) -> 2 WMMAs per K-step sharing one A fragment.
// B tile is transposed into LDS (stride 40 halves -> aligned b128 reads).
// =====================================================================
template <int BF16>
__global__ void __launch_bounds__(256)
gemm_wmma_kernel(const _Float16* __restrict__ A, int Kpad, int Mvalid,
                 const void* __restrict__ Bsrc, int bRow, int Kvalid,
                 float* __restrict__ Out, int oRow,
                 const float* __restrict__ bias, int flags)
{
    __shared__ __align__(16) _Float16 Bt[128 * 40];

    const int tid  = threadIdx.x;
    const int n0   = blockIdx.x * 128;
    const int m0   = blockIdx.y * 32;
    const int w    = tid >> 5;
    const int lane = tid & 31;
    const int tr   = w & 1;        // wave tile row (0..1)
    const int tc   = w >> 1;       // wave tile col (0..3), 32 cols each
    const int ml   = lane & 15;
    const int hiH  = (lane >= 16); // lane half
    const int kbA  = hiH ? 8 : 0;  // A frag K base (ISA 16-bit A 16x32 layout)
    const int kbB  = hiH ? 8 : 0;  // B frag K base (same K order as A)

    f32x8 acc0, acc1;
    #pragma unroll
    for (int i = 0; i < 8; ++i) { acc0[i] = 0.f; acc1[i] = 0.f; }

    const int mbase = m0 + tr * 16;
    const _Float16* Arow = A + (long)(mbase + ml) * Kpad;

    for (int k0 = 0; k0 < Kpad; k0 += 32) {
        // ---- cooperative fill of transposed B tile: 32 K x 128 N ----
        {
            int kk = tid >> 3;          // 0..31
            int nn = (tid & 7) * 16;    // 0,16,...,112
            int gk = k0 + kk;
            _Float16 tmp[16];
            if (gk < Kvalid) {
                long base = (long)gk * bRow + (n0 + nn);
                if (BF16) {
                    const f16x8* p = (const f16x8*)((const _Float16*)Bsrc + base);
                    f16x8 a = p[0], b = p[1];
                    #pragma unroll
                    for (int j = 0; j < 8; ++j) { tmp[j] = a[j]; tmp[8 + j] = b[j]; }
                } else {
                    const float4* p = (const float4*)((const float*)Bsrc + base);
                    float4 a = p[0], b = p[1], c = p[2], d = p[3];
                    tmp[0]  = (_Float16)a.x; tmp[1]  = (_Float16)a.y;
                    tmp[2]  = (_Float16)a.z; tmp[3]  = (_Float16)a.w;
                    tmp[4]  = (_Float16)b.x; tmp[5]  = (_Float16)b.y;
                    tmp[6]  = (_Float16)b.z; tmp[7]  = (_Float16)b.w;
                    tmp[8]  = (_Float16)c.x; tmp[9]  = (_Float16)c.y;
                    tmp[10] = (_Float16)c.z; tmp[11] = (_Float16)c.w;
                    tmp[12] = (_Float16)d.x; tmp[13] = (_Float16)d.y;
                    tmp[14] = (_Float16)d.z; tmp[15] = (_Float16)d.w;
                }
            } else {
                #pragma unroll
                for (int j = 0; j < 16; ++j) tmp[j] = (_Float16)0.f;
            }
            #pragma unroll
            for (int j = 0; j < 16; ++j) Bt[(nn + j) * 40 + kk] = tmp[j];
        }
        __syncthreads();

        FragU ua, ub0, ub1;
        // A fragment (shared by both N tiles): halves 0..7 = K kbA..kbA+7,
        // halves 8..15 = K kbA+16..kbA+23
        ua.h[0] = *(const f16x8*)(Arow + k0 + kbA);
        ua.h[1] = *(const f16x8*)(Arow + k0 + kbA + 16);
        // B fragments: lane ml holds column; K kbB..kbB+7, kbB+16..kbB+23
        const _Float16* bp0 = &Bt[(tc * 32 + ml) * 40];
        const _Float16* bp1 = &Bt[(tc * 32 + 16 + ml) * 40];
        ub0.h[0] = *(const f16x8*)(bp0 + kbB);
        ub0.h[1] = *(const f16x8*)(bp0 + kbB + 16);
        ub1.h[0] = *(const f16x8*)(bp1 + kbB);
        ub1.h[1] = *(const f16x8*)(bp1 + kbB + 16);

        acc0 = __builtin_amdgcn_wmma_f32_16x16x32_f16(
                   false, ua.v, false, ub0.v, (short)0, acc0, false, false);
        acc1 = __builtin_amdgcn_wmma_f32_16x16x32_f16(
                   false, ua.v, false, ub1.v, (short)0, acc1, false, false);
        __syncthreads();
    }

    const int nA   = n0 + tc * 32 + ml;
    const int mofs = hiH ? 8 : 0;
    #pragma unroll
    for (int v = 0; v < 8; ++v) {
        int m = mbase + v + mofs;
        if (m < Mvalid) {
            float bsv = (flags & 1) ? bias[m] : 0.f;
            Out[(long)m * oRow + nA]      = acc0[v] + bsv;
            Out[(long)m * oRow + nA + 16] = acc1[v] + bsv;
        }
    }
}

// =====================================================================
// Weight packing: f32 -> f16, zero padded to Mpad x Kpad
// =====================================================================
__global__ void pack_a_kernel(const float* __restrict__ src, _Float16* __restrict__ dst,
                              int Mpad, int Mvalid, int Kpad, int Kvalid, int srcRow)
{
    long total = (long)Mpad * Kpad;
    for (long idx = (long)blockIdx.x * blockDim.x + threadIdx.x; idx < total;
         idx += (long)gridDim.x * blockDim.x) {
        int k = (int)(idx % Kpad);
        int m = (int)(idx / Kpad);
        _Float16 v = (_Float16)0.f;
        if (m < Mvalid && k < Kvalid) v = (_Float16)src[(long)m * srcRow + k];
        dst[idx] = v;
    }
}

__global__ void negexp_kernel(const float* __restrict__ src, float* __restrict__ dst, int n)
{
    int i = blockIdx.x * blockDim.x + threadIdx.x;
    if (i < n) dst[i] = -__expf(src[i]);
}

// =====================================================================
// LayerNorm over channel dim, generic strides. Element (b,c,hw):
//   src + b*sB + c*sC + hw.  flags: 1=relu, 2=add aux (skip), 4=mul silu(aux)
// =====================================================================
__global__ void ln2d_kernel(const float* __restrict__ src, long sB, long sC,
                            float* __restrict__ dst, long dB, long dC,
                            const float* __restrict__ g, const float* __restrict__ be,
                            const float* __restrict__ aux, long aB, long aC,
                            int C, int flags)
{
    int idx = blockIdx.x * blockDim.x + threadIdx.x;
    if (idx >= NBAT * LHW) return;
    int b = idx / LHW, hw = idx % LHW;
    const float* sp = src + (long)b * sB + hw;
    float s = 0.f, ss = 0.f;
    for (int c = 0; c < C; ++c) { float v = sp[(long)c * sC]; s += v; ss += v * v; }
    float mu = s / C;
    float var = ss / C - mu * mu;
    float rstd = rsqrtf(var + 1e-6f);
    float* dp = dst + (long)b * dB + hw;
    const float* ap = aux ? aux + (long)b * aB + hw : nullptr;
    for (int c = 0; c < C; ++c) {
        float v = (sp[(long)c * sC] - mu) * rstd * g[c] + be[c];
        if (flags & 1) v = fmaxf(v, 0.f);
        if (flags & 2) v += ap[(long)c * aC];
        if (flags & 4) { float z = ap[(long)c * aC]; v *= z * sigmoidf_(z); }
        dp[(long)c * dC] = v;
    }
}

// =====================================================================
// im2col for 3x3 SAME conv -> f16 matrix [C*9][NCOL]
// =====================================================================
__global__ void im2col9_kernel(const float* __restrict__ src, long sB, long sC,
                               _Float16* __restrict__ dst, int C)
{
    long total = (long)C * 9 * NCOL;
    for (long idx = (long)blockIdx.x * blockDim.x + threadIdx.x; idx < total;
         idx += (long)gridDim.x * blockDim.x) {
        int  n   = (int)(idx % NCOL);
        long r   = idx / NCOL;
        int  tap = (int)(r % 9);
        int  c   = (int)(r / 9);
        int  b   = n / LHW, hw = n % LHW;
        int  h   = hw / HH + tap / 3 - 1;
        int  w   = hw % HH + tap % 3 - 1;
        float v  = 0.f;
        if (h >= 0 && h < HH && w >= 0 && w < HH)
            v = src[(long)b * sB + (long)c * sC + h * HH + w];
        dst[idx] = (_Float16)v;
    }
}

// =====================================================================
// depthwise 3x3 SAME conv + silu on xz rows [0..DI), CM layout [d][NCOL]
// =====================================================================
__global__ void dwconv_silu_kernel(const float* __restrict__ xz,
                                   const float* __restrict__ wgt,
                                   const float* __restrict__ bcv,
                                   float* __restrict__ xp)
{
    long total = (long)DI * NCOL;
    for (long idx = (long)blockIdx.x * blockDim.x + threadIdx.x; idx < total;
         idx += (long)gridDim.x * blockDim.x) {
        int n = (int)(idx % NCOL);
        int d = (int)(idx / NCOL);
        int b = n / LHW, hw = n % LHW, h = hw / HH, w = hw % HH;
        const float* srow = xz + (long)d * NCOL + (long)b * LHW;
        const float* wp   = wgt + d * 9;
        float acc = bcv[d];
        #pragma unroll
        for (int t = 0; t < 9; ++t) {
            int hh = h + t / 3 - 1, ww = w + t % 3 - 1;
            if (hh >= 0 && hh < HH && ww >= 0 && ww < HH)
                acc += wp[t] * srow[hh * HH + ww];
        }
        xp[idx] = acc * sigmoidf_(acc);
    }
}

// =====================================================================
// xs[k][d][b*L+l] (f16) = xp at scan-order-k position
// k0: l ; k1: transpose (l=w*H+h) ; k2: reverse ; k3: transpose+reverse
// =====================================================================
__global__ void build_xs_kernel(const float* __restrict__ xp, _Float16* __restrict__ xs)
{
    long total = (long)KDIR * DI * NCOL;
    for (long idx = (long)blockIdx.x * blockDim.x + threadIdx.x; idx < total;
         idx += (long)gridDim.x * blockDim.x) {
        int n = (int)(idx % NCOL);
        int d = (int)((idx / NCOL) % DI);
        int k = (int)(idx / ((long)DI * NCOL));
        int b = n / LHW, l = n % LHW;
        int src;
        if (k == 0)      src = l;
        else if (k == 1) { int h = l % HH, w = l / HH; src = h * HH + w; }
        else if (k == 2) src = (LHW - 1) - l;
        else             { int lr = (LHW - 1) - l; int h = lr % HH, w = lr / HH; src = h * HH + w; }
        xs[idx] = (_Float16)xp[(long)d * NCOL + (long)b * LHW + src];
    }
}

// softplus(x + dt_bias[k][d]) in place on dt [K][DI][NCOL]
__global__ void softplus_kernel(float* __restrict__ dt, const float* __restrict__ dtb)
{
    long total = (long)KDIR * DI * NCOL;
    for (long idx = (long)blockIdx.x * blockDim.x + threadIdx.x; idx < total;
         idx += (long)gridDim.x * blockDim.x) {
        int d = (int)((idx / NCOL) % DI);
        int k = (int)(idx / ((long)DI * NCOL));
        float x = dt[idx] + dtb[k * DI + d];
        dt[idx] = (x > 20.f) ? x : __logf(1.f + __expf(x));
    }
}

// =====================================================================
// Selective scan: 16-lane group per (b,k,d); lane = state index s.
// h_l = h_{l-1}*exp(dt*A) + dt*u*B ; y = sum_s h*C + D*u
// =====================================================================
__global__ void __launch_bounds__(256)
scan_kernel(const float* __restrict__ dt, const float* __restrict__ xdbl,
            const _Float16* __restrict__ xs, const float* __restrict__ Aexp,
            const float* __restrict__ Dp, float* __restrict__ ys)
{
    int tid = threadIdx.x;
    int s   = tid & 15;
    int g   = blockIdx.x * 16 + (tid >> 4);   // 0..6143
    int d   = g % DI;
    int k   = (g / DI) & 3;
    int b   = g / (DI * KDIR);

    const float*    dtp = dt   + ((long)(k * DI + d)) * NCOL + (long)b * LHW;
    const float*    bsp = xdbl + ((long)(k * 64 + 6 + s))  * NCOL + (long)b * LHW;
    const float*    csp = xdbl + ((long)(k * 64 + 22 + s)) * NCOL + (long)b * LHW;
    const _Float16* up  = xs   + ((long)(k * DI + d)) * NCOL + (long)b * LHW;
    float*          yp  = ys   + ((long)(k * DI + d)) * NCOL + (long)b * LHW;
    const float Ak = Aexp[((k * DI + d) << 4) + s];
    const float Dv = Dp[k * DI + d];

    float h = 0.f;
    for (int l = 0; l < LHW; ++l) {
        float dtv = dtp[l];
        float u   = (float)up[l];
        h = h * __expf(dtv * Ak) + dtv * u * bsp[l];
        float part = h * csp[l];
        part += __shfl_xor(part, 1, 32);
        part += __shfl_xor(part, 2, 32);
        part += __shfl_xor(part, 4, 32);
        part += __shfl_xor(part, 8, 32);
        if (s == 0) yp[l] = part + Dv * u;
    }
}

// combine 4 scan directions -> yc [DI][NCOL]
__global__ void combine_kernel(const float* __restrict__ ys, float* __restrict__ yc)
{
    long total = (long)DI * NCOL;
    for (long idx = (long)blockIdx.x * blockDim.x + threadIdx.x; idx < total;
         idx += (long)gridDim.x * blockDim.x) {
        int n = (int)(idx % NCOL);
        int d = (int)(idx / NCOL);
        int b = n / LHW, l = n % LHW;
        int h = l / HH, w = l % HH;
        int lw = w * HH + h;
        long base = (long)b * LHW;
        const float* y0 = ys + ((long)(0 * DI + d)) * NCOL + base;
        const float* y1 = ys + ((long)(1 * DI + d)) * NCOL + base;
        const float* y2 = ys + ((long)(2 * DI + d)) * NCOL + base;
        const float* y3 = ys + ((long)(3 * DI + d)) * NCOL + base;
        yc[idx] = y0[l] + y2[(LHW - 1) - l] + y1[lw] + y3[(LHW - 1) - lw];
    }
}

// per-pixel KL on CM layout [c][b*LHW+hw]
__global__ void cal_kl_kernel(const float* __restrict__ p, const float* __restrict__ q,
                              float* __restrict__ out)
{
    int idx = blockIdx.x * blockDim.x + threadIdx.x;
    if (idx >= NBAT * LHW) return;
    int b = idx / LHW, hw = idx % LHW;
    long ofs = (long)b * LHW + hw;
    const float* pp = p + ofs;
    const float* qp = q + ofs;
    float mp = -1e30f, mq = -1e30f;
    for (int c = 0; c < C96; ++c) {
        mp = fmaxf(mp, pp[(long)c * NCOL]);
        mq = fmaxf(mq, qp[(long)c * NCOL]);
    }
    float sp = 0.f, sq = 0.f;
    for (int c = 0; c < C96; ++c) {
        sp += __expf(pp[(long)c * NCOL] - mp);
        sq += __expf(qp[(long)c * NCOL] - mq);
    }
    float lzp = mp + __logf(sp), lzq = mq + __logf(sq);
    for (int c = 0; c < C96; ++c) {
        float lp = pp[(long)c * NCOL] - lzp;
        float lq = qp[(long)c * NCOL] - lzq;
        out[(long)c * NCOL + ofs] = __expf(lp) * (lp - lq);
    }
}

// previous = current * exp(grouped_conv3x3(sigma) + ci_b), groups=3
__global__ void ci_prev_kernel(const float* __restrict__ sigma, const float* __restrict__ w,
                               const float* __restrict__ bias, const float* __restrict__ cur,
                               float* __restrict__ prev)
{
    long total = (long)NBAT * C96 * LHW;
    for (long idx = (long)blockIdx.x * blockDim.x + threadIdx.x; idx < total;
         idx += (long)gridDim.x * blockDim.x) {
        int hw = (int)(idx % LHW);
        int c  = (int)((idx / LHW) % C96);
        int b  = (int)(idx / ((long)C96 * LHW));
        int gi = c / 32;
        int h = hw / HH, ww = hw % HH;
        float acc = bias[c];
        for (int ci = 0; ci < 32; ++ci) {
            const float* sp = sigma + ((long)(b * C96 + gi * 32 + ci)) * LHW;
            const float* wp = w + ((long)c * 32 + ci) * 9;
            #pragma unroll
            for (int t = 0; t < 9; ++t) {
                int hh = h + t / 3 - 1, wq = ww + t % 3 - 1;
                if (hh >= 0 && hh < HH && wq >= 0 && wq < HH)
                    acc += wp[t] * sp[hh * HH + wq];
            }
        }
        long o = (long)c * NCOL + (long)b * LHW + hw;
        prev[o] = cur[o] * __expf(acc);
    }
}

// next = cur * exp(ss + si) + bb
__global__ void update_kernel(const float* __restrict__ cur, const float* __restrict__ ss,
                              const float* __restrict__ si, const float* __restrict__ bb,
                              float* __restrict__ nxt)
{
    long total = (long)C96 * NCOL;
    for (long idx = (long)blockIdx.x * blockDim.x + threadIdx.x; idx < total;
         idx += (long)gridDim.x * blockDim.x)
        nxt[idx] = cur[idx] * __expf(ss[idx] + si[idx]) + bb[idx];
}

// CM [c][b*LHW+hw] -> NCHW output
__global__ void copyout_kernel(const float* __restrict__ cur, float* __restrict__ out)
{
    long total = (long)NBAT * C96 * LHW;
    for (long idx = (long)blockIdx.x * blockDim.x + threadIdx.x; idx < total;
         idx += (long)gridDim.x * blockDim.x) {
        int hw = (int)(idx % LHW);
        int c  = (int)((idx / LHW) % C96);
        int b  = (int)(idx / ((long)C96 * LHW));
        out[idx] = cur[(long)c * NCOL + (long)b * LHW + hw];
    }
}

// =====================================================================
// Host side
// =====================================================================
static inline int cdiv_i(long a, int b) { return (int)((a + b - 1) / b); }

struct SS2DWeights {
    const float *conv_w, *conv_b, *dt_bias, *D, *on_g, *on_b;
    const _Float16 *inP, *xP, *dtP, *outP;
    const float *Aexp;
};

struct Scratch {
    float* XZ; float* XP; _Float16* XS; float* XDBL;
    float* DT; float* YS; float* YC; float* YN;
};

static void run_ss2d(const float* x, const SS2DWeights& w, const Scratch& s,
                     float* out, hipStream_t st)
{
    // in_proj: [384x96] * [96 x NCOL]
    gemm_wmma_kernel<0><<<dim3(NCOL / 128, 384 / 32), 256, 0, st>>>(
        w.inP, 96, 384, x, NCOL, 96, s.XZ, NCOL, nullptr, 0);

    dwconv_silu_kernel<<<cdiv_i((long)DI * NCOL, 256), 256, 0, st>>>(
        s.XZ, w.conv_w, w.conv_b, s.XP);

    build_xs_kernel<<<cdiv_i((long)KDIR * DI * NCOL, 256), 256, 0, st>>>(s.XP, s.XS);

    for (int k = 0; k < KDIR; ++k)
        gemm_wmma_kernel<1><<<dim3(NCOL / 128, 64 / 32), 256, 0, st>>>(
            w.xP + (long)k * 64 * 192, 192, 38,
            s.XS + (long)k * DI * NCOL, NCOL, 192,
            s.XDBL + (long)k * 64 * NCOL, NCOL, nullptr, 0);

    for (int k = 0; k < KDIR; ++k)
        gemm_wmma_kernel<0><<<dim3(NCOL / 128, 192 / 32), 256, 0, st>>>(
            w.dtP + (long)k * 192 * 32, 32, 192,
            s.XDBL + (long)k * 64 * NCOL, NCOL, 6,
            s.DT + (long)k * DI * NCOL, NCOL, nullptr, 0);

    softplus_kernel<<<cdiv_i((long)KDIR * DI * NCOL, 256), 256, 0, st>>>(s.DT, w.dt_bias);

    scan_kernel<<<384, 256, 0, st>>>(s.DT, s.XDBL, s.XS, w.Aexp, w.D, s.YS);

    combine_kernel<<<cdiv_i((long)DI * NCOL, 256), 256, 0, st>>>(s.YS, s.YC);

    // ln over DI * silu(z); z = xz rows DI..2DI
    ln2d_kernel<<<NCOL / 256, 256, 0, st>>>(
        s.YC, LHW, NCOL, s.YN, LHW, NCOL, w.on_g, w.on_b,
        s.XZ + (long)DI * NCOL, LHW, NCOL, DI, 4);

    // out_proj: [96x192] * [192 x NCOL]
    gemm_wmma_kernel<0><<<dim3(NCOL / 128, 96 / 32), 256, 0, st>>>(
        w.outP, 192, 96, s.YN, NCOL, 192, out, NCOL, nullptr, 0);
}

static void pack_module(void* const* d_in, int base,
                        _Float16* inP, _Float16* xP, _Float16* dtP, _Float16* outP,
                        float* Aexp, hipStream_t st)
{
    const float* in_proj  = (const float*)d_in[base + 0];
    const float* x_proj   = (const float*)d_in[base + 3];
    const float* dt_proj  = (const float*)d_in[base + 4];
    const float* A_log    = (const float*)d_in[base + 6];
    const float* out_proj = (const float*)d_in[base + 10];
    pack_a_kernel<<<cdiv_i(384 * 96, 256), 256, 0, st>>>(in_proj, inP, 384, 384, 96, 96, 96);
    for (int k = 0; k < KDIR; ++k) {
        pack_a_kernel<<<cdiv_i(64 * 192, 256), 256, 0, st>>>(
            x_proj + (long)k * 38 * 192, xP + (long)k * 64 * 192, 64, 38, 192, 192, 192);
        pack_a_kernel<<<cdiv_i(192 * 32, 256), 256, 0, st>>>(
            dt_proj + (long)k * 192 * 6, dtP + (long)k * 192 * 32, 192, 192, 32, 6, 6);
    }
    pack_a_kernel<<<cdiv_i(96 * 192, 256), 256, 0, st>>>(out_proj, outP, 96, 96, 192, 192, 192);
    negexp_kernel<<<cdiv_i(KDIR * DI * DS16, 256), 256, 0, st>>>(A_log, Aexp, KDIR * DI * DS16);
}

extern "C" void kernel_launch(void* const* d_in, const int* in_sizes, int n_in,
                              void* d_out, int out_size, void* d_ws, size_t ws_size,
                              hipStream_t stream)
{
    char* ws = (char*)d_ws;
    size_t off = 0;
    auto take = [&](size_t bytes) -> char* {
        char* p = ws + off;
        off += (bytes + 255) & ~(size_t)255;
        return p;
    };

    const size_t SMALLB = (size_t)C96 * NCOL * 4;   // one CM f32 activation
    float* CUR[3]; for (int i = 0; i < 3; ++i) CUR[i] = (float*)take(SMALLB);
    float* IMG = (float*)take(SMALLB);
    float* T1  = (float*)take(SMALLB);
    float* T2  = (float*)take(SMALLB);
    float* FE  = (float*)take(SMALLB);
    float* SI  = (float*)take(SMALLB);
    float* SS  = (float*)take(SMALLB);
    float* BB  = (float*)take(SMALLB);
    float* KL  = (float*)take(SMALLB);

    Scratch sc;
    sc.XZ   = (float*)take((size_t)2 * DI * NCOL * 4);
    sc.XP   = (float*)take((size_t)DI * NCOL * 4);
    sc.XS   = (_Float16*)take((size_t)KDIR * DI * NCOL * 2);
    sc.XDBL = (float*)take((size_t)KDIR * 64 * NCOL * 4);
    sc.DT   = (float*)take((size_t)KDIR * DI * NCOL * 4);
    sc.YS   = (float*)take((size_t)KDIR * DI * NCOL * 4);
    // regions dead after the scan get aliased into DT:
    sc.YC   = sc.DT;
    sc.YN   = sc.DT + (size_t)DI * NCOL;
    _Float16* IC = (_Float16*)sc.DT;                // im2col (res_block only)

    // packed weights
    _Float16* Wc1[SEQLEN]; _Float16* Wc2[SEQLEN];
    for (int i = 0; i < SEQLEN; ++i) {
        Wc1[i] = (_Float16*)take((size_t)96 * 864 * 2);
        Wc2[i] = (_Float16*)take((size_t)96 * 864 * 2);
    }
    _Float16 *inP[3], *xP[3], *dtP[3], *outP[3]; float* Aexp[3];
    for (int m = 0; m < 3; ++m) {
        inP[m]  = (_Float16*)take((size_t)384 * 96 * 2);
        xP[m]   = (_Float16*)take((size_t)KDIR * 64 * 192 * 2);
        dtP[m]  = (_Float16*)take((size_t)KDIR * 192 * 32 * 2);
        outP[m] = (_Float16*)take((size_t)96 * 192 * 2);
        Aexp[m] = (float*)take((size_t)KDIR * DI * DS16 * 4);
    }

    // -------- inputs (setup_inputs dict order, params recursively) --------
    const float* image_seq = (const float*)d_in[0];
    const float* dzone     = (const float*)d_in[1];
    const float* sigma     = (const float*)d_in[2];
    const float* nd_g      = (const float*)d_in[3];
    const float* nd_b      = (const float*)d_in[4];
    const float* ci_w      = (const float*)d_in[5];
    const float* ci_b      = (const float*)d_in[6];
    const int BLK0 = 7;                 // 4 blocks x 10 tensors
    const int MOD[3] = { 47, 58, 69 };  // m_state, m_image, m_bias

    // -------- pack weights --------
    for (int i = 0; i < SEQLEN; ++i) {
        const float* w1 = (const float*)d_in[BLK0 + i * 10 + 2];
        const float* w2 = (const float*)d_in[BLK0 + i * 10 + 6];
        pack_a_kernel<<<cdiv_i(96 * 864, 256), 256, 0, stream>>>(w1, Wc1[i], 96, 96, 864, 864, 864);
        pack_a_kernel<<<cdiv_i(96 * 864, 256), 256, 0, stream>>>(w2, Wc2[i], 96, 96, 864, 864, 864);
    }
    SS2DWeights sw[3];
    for (int m = 0; m < 3; ++m) {
        pack_module(d_in, MOD[m], inP[m], xP[m], dtP[m], outP[m], Aexp[m], stream);
        sw[m].conv_w  = (const float*)d_in[MOD[m] + 1];
        sw[m].conv_b  = (const float*)d_in[MOD[m] + 2];
        sw[m].dt_bias = (const float*)d_in[MOD[m] + 5];
        sw[m].D       = (const float*)d_in[MOD[m] + 7];
        sw[m].on_g    = (const float*)d_in[MOD[m] + 8];
        sw[m].on_b    = (const float*)d_in[MOD[m] + 9];
        sw[m].inP = inP[m]; sw[m].xP = xP[m]; sw[m].dtP = dtP[m]; sw[m].outP = outP[m];
        sw[m].Aexp = Aexp[m];
    }
    const SS2DWeights& m_state = sw[0];
    const SS2DWeights& m_image = sw[1];
    const SS2DWeights& m_bias  = sw[2];

    // -------- init: current = ln(difficult_zone); previous = cur*exp(conv_i(sigma)) --------
    float* curp = CUR[0]; float* prevp = CUR[1]; float* nextp = CUR[2];
    ln2d_kernel<<<NCOL / 256, 256, 0, stream>>>(
        dzone, (long)C96 * LHW, LHW, curp, LHW, NCOL,
        nd_g, nd_b, nullptr, 0, 0, C96, 0);
    ci_prev_kernel<<<cdiv_i((long)NBAT * C96 * LHW, 256), 256, 0, stream>>>(
        sigma, ci_w, ci_b, curp, prevp);

    // -------- main loop --------
    for (int i = 0; i < SEQLEN; ++i) {
        int pb = BLK0 + i * 10;
        const float* ni_g = (const float*)d_in[pb + 0];
        const float* ni_b = (const float*)d_in[pb + 1];
        const float* b1   = (const float*)d_in[pb + 3];
        const float* g1   = (const float*)d_in[pb + 4];
        const float* be1  = (const float*)d_in[pb + 5];
        const float* b2   = (const float*)d_in[pb + 7];
        const float* g2   = (const float*)d_in[pb + 8];
        const float* be2  = (const float*)d_in[pb + 9];

        // image = ln(image_sequence[:, i]) -> CM
        ln2d_kernel<<<NCOL / 256, 256, 0, stream>>>(
            image_seq + (size_t)i * C96 * LHW, (long)SEQLEN * C96 * LHW, LHW,
            IMG, LHW, NCOL, ni_g, ni_b, nullptr, 0, 0, C96, 0);

        // res_block: conv1 -> ln+relu -> conv2 -> ln + skip
        im2col9_kernel<<<cdiv_i((long)C96 * 9 * NCOL, 256), 256, 0, stream>>>(
            IMG, LHW, NCOL, IC, C96);
        gemm_wmma_kernel<1><<<dim3(NCOL / 128, 96 / 32), 256, 0, stream>>>(
            Wc1[i], 864, 96, IC, NCOL, 864, T1, NCOL, b1, 1);
        ln2d_kernel<<<NCOL / 256, 256, 0, stream>>>(
            T1, LHW, NCOL, T2, LHW, NCOL, g1, be1, nullptr, 0, 0, C96, 1);
        im2col9_kernel<<<cdiv_i((long)C96 * 9 * NCOL, 256), 256, 0, stream>>>(
            T2, LHW, NCOL, IC, C96);
        gemm_wmma_kernel<1><<<dim3(NCOL / 128, 96 / 32), 256, 0, stream>>>(
            Wc2[i], 864, 96, IC, NCOL, 864, T1, NCOL, b2, 1);
        ln2d_kernel<<<NCOL / 256, 256, 0, stream>>>(
            T1, LHW, NCOL, FE, LHW, NCOL, g2, be2, IMG, LHW, NCOL, C96, 2);

        // three SS2D branches
        run_ss2d(FE, m_image, sc, SI, stream);
        cal_kl_kernel<<<NCOL / 256, 256, 0, stream>>>(curp, prevp, KL);
        run_ss2d(KL, m_state, sc, SS, stream);
        run_ss2d(curp, m_bias, sc, BB, stream);

        // state update + rotate (previous = old current)
        update_kernel<<<cdiv_i((long)C96 * NCOL, 256), 256, 0, stream>>>(
            curp, SS, SI, BB, nextp);
        float* old_prev = prevp;
        prevp = curp; curp = nextp; nextp = old_prev;
    }

    copyout_kernel<<<cdiv_i((long)NBAT * C96 * LHW, 256), 256, 0, stream>>>(
        curp, (float*)d_out);
}
